// DTIConvGraph3Test_6373731467775
// MI455X (gfx1250) — matrix operand order repaired and checked by reading, not compiled, not run on hardware
//
#include <hip/hip_runtime.h>
#include <hip/hip_bf16.h>

typedef _Float16 h8   __attribute__((ext_vector_type(8)));
typedef _Float16 v16h __attribute__((ext_vector_type(16)));
typedef float    v8f  __attribute__((ext_vector_type(8)));
typedef float    f4   __attribute__((ext_vector_type(4)));

#define NEG_SLOPE 0.01f
#define LDSS 136   // padded LDS row stride in halves (272B -> 4-bank shift/row, conflict-free)
#define TPB 4      // tiles of 16 edges per block

// ---------------------------------------------------------------------------
// Prep: fuse W1/W2 into two f16 128x128 matrices + rank-1 col + fused bias.
// WA[n][k] = W1[n][k]     + W2[n][k]
// WB[n][k] = W1[n][128+k] + W2[n][k]
// wc[n]    = W1[n][256] ; bias[n] = b1[n]+b2[n]
// ---------------------------------------------------------------------------
__global__ __launch_bounds__(256) void prep_weights(
    const float* __restrict__ W1, const float* __restrict__ b1,
    const float* __restrict__ W2, const float* __restrict__ b2,
    _Float16* __restrict__ WA, _Float16* __restrict__ WB,
    float* __restrict__ wc, float* __restrict__ bias)
{
  int idx = blockIdx.x * 256 + threadIdx.x;
  if (idx < 128 * 128) {
    int n = idx >> 7, k = idx & 127;
    float w2 = W2[n * 128 + k];
    WA[idx] = (_Float16)(W1[n * 257 + k] + w2);
    WB[idx] = (_Float16)(W1[n * 257 + 128 + k] + w2);
  }
  if (idx < 128) {
    wc[idx]   = W1[idx * 257 + 256];
    bias[idx] = b1[idx] + b2[idx];
  }
}

// ---------------------------------------------------------------------------
// Main: per 16-edge tile, Out = Hs@WA + Hd@WB + bond*wc + bias, leaky-ReLU.
// 8 waves/block, each wave owns a 16-wide N slice; 8 WMMAs per tile per wave.
// ---------------------------------------------------------------------------
__global__ __launch_bounds__(256) void edge_mlp(
    const float* __restrict__ atom, const float* __restrict__ bond,
    const int* __restrict__ src, const int* __restrict__ dst,
    const _Float16* __restrict__ WA, const _Float16* __restrict__ WB,
    const float* __restrict__ wc, const float* __restrict__ bias,
    float* __restrict__ out, int E)
{
  __shared__ _Float16 shs[16 * LDSS];
  __shared__ _Float16 shd[16 * LDSS];
  __shared__ float    sbond[16];

  const int t    = threadIdx.x;
  const int lane = t & 31;
  const int wave = t >> 5;
  const int row  = lane & 15;
  const int hi   = lane >> 4;
  const int n    = wave * 16 + row;   // output column owned by this lane

  // Hoist weight B-fragments into registers (amortized over TPB tiles).
  // B fragment, contiguous k order (same as A): lane = N, VGPR j holds K = hi*16 + 2j,2j+1.
  v16h fA[4], fB[4];
#pragma unroll
  for (int kb = 0; kb < 4; ++kb) {
    const h8* pa = (const h8*)(WA + (size_t)n * 128 + kb * 32 + hi * 16);
    const h8* pb = (const h8*)(WB + (size_t)n * 128 + kb * 32 + hi * 16);
    h8 a0 = pa[0], a1 = pa[1], b0 = pb[0], b1 = pb[1];
#pragma unroll
    for (int i = 0; i < 8; ++i) {
      fA[kb][i] = a0[i]; fA[kb][8 + i] = a1[i];
      fB[kb][i] = b0[i]; fB[kb][8 + i] = b1[i];
    }
  }
  const float biasn = bias[n];
  const float wcn   = wc[n];

  const int r  = t >> 4;          // edge row this thread stages
  const int c8 = (t & 15) * 8;    // 8-float column chunk

#pragma unroll 1
  for (int tile = 0; tile < TPB; ++tile) {
    const int tb = blockIdx.x * (TPB * 16) + tile * 16;
    if (tb >= E) break;

    // ---- stage 16x128 gathered src/dst features into LDS (f32 -> f16) ----
    int e = tb + r; e = e < E ? e : E - 1;
    const size_t so   = (size_t)src[e] * 128 + c8;
    const size_t dofs = (size_t)dst[e] * 128 + c8;
    f4 s0 = *(const f4*)(atom + so);
    f4 s1 = *(const f4*)(atom + so + 4);
    f4 d0 = *(const f4*)(atom + dofs);
    f4 d1 = *(const f4*)(atom + dofs + 4);
    h8 hsv, hdv;
#pragma unroll
    for (int i = 0; i < 4; ++i) {
      hsv[i] = (_Float16)s0[i]; hsv[4 + i] = (_Float16)s1[i];
      hdv[i] = (_Float16)d0[i]; hdv[4 + i] = (_Float16)d1[i];
    }
    *(h8*)(&shs[r * LDSS + c8]) = hsv;
    *(h8*)(&shd[r * LDSS + c8]) = hdv;
    if (t < 16) {
      int eb = tb + t;
      sbond[t] = bond[eb < E ? eb : E - 1];
    }
    __syncthreads();

    // ---- load ALL A-fragments first (16x ds_load_b128), then WMMA chain ----
    // A fragment, contiguous k order (same as B): lane row = M, VGPR0-3 = K[hi*16..+8),
    // VGPR4-7 = K[hi*16+8..+8)  -> two contiguous 16B LDS reads per frag.
    v16h aS[4], aD[4];
#pragma unroll
    for (int kb = 0; kb < 4; ++kb) {
      const int k0 = kb * 32;
      h8 sl = *(const h8*)(&shs[row * LDSS + k0 + hi * 16]);
      h8 sh = *(const h8*)(&shs[row * LDSS + k0 + hi * 16 + 8]);
      h8 dl = *(const h8*)(&shd[row * LDSS + k0 + hi * 16]);
      h8 dh = *(const h8*)(&shd[row * LDSS + k0 + hi * 16 + 8]);
#pragma unroll
      for (int i = 0; i < 8; ++i) {
        aS[kb][i] = sl[i]; aS[kb][8 + i] = sh[i];
        aD[kb][i] = dl[i]; aD[kb][8 + i] = dh[i];
      }
    }
    // bond values for the 8 rows this lane touches (two 16B LDS loads)
    f4 bnd0 = *(const f4*)(&sbond[hi * 8]);
    f4 bnd1 = *(const f4*)(&sbond[hi * 8 + 4]);

    v8f acc = {};
#pragma unroll
    for (int kb = 0; kb < 4; ++kb) {
      acc = __builtin_amdgcn_wmma_f32_16x16x32_f16(false, aS[kb], false, fA[kb],
                                                   (short)0, acc, false, false);
      acc = __builtin_amdgcn_wmma_f32_16x16x32_f16(false, aD[kb], false, fB[kb],
                                                   (short)0, acc, false, false);
    }

    // ---- epilogue: +bias +bond*wc, leaky ReLU, store ----
    // C/D layout: VGPR v -> M = v + 8*hi, N = lane&15.
    float res[8];
#pragma unroll
    for (int v = 0; v < 8; ++v) {
      const float bd = (v < 4) ? bnd0[v & 3] : bnd1[v & 3];
      float val = acc[v] + biasn + bd * wcn;
      res[v] = val > 0.0f ? val : val * NEG_SLOPE;
    }
    float* outp = out + (size_t)(tb + hi * 8) * 128 + n;
    if (tb + 16 <= E) {           // uniform fast path: no exec juggling
#pragma unroll
      for (int v = 0; v < 8; ++v) outp[(size_t)v * 128] = res[v];
    } else {
#pragma unroll
      for (int v = 0; v < 8; ++v)
        if (tb + hi * 8 + v < E) outp[(size_t)v * 128] = res[v];
    }
    __syncthreads();
  }
}

// ---------------------------------------------------------------------------
extern "C" void kernel_launch(void* const* d_in, const int* in_sizes, int n_in,
                              void* d_out, int out_size, void* d_ws, size_t ws_size,
                              hipStream_t stream) {
  const float* atom = (const float*)d_in[0];
  const float* bond = (const float*)d_in[1];
  const int*   src  = (const int*)d_in[2];
  const int*   dst  = (const int*)d_in[3];
  const float* W1   = (const float*)d_in[4];
  const float* b1   = (const float*)d_in[5];
  const float* W2   = (const float*)d_in[6];
  const float* b2   = (const float*)d_in[7];
  float* out = (float*)d_out;
  const int E = in_sizes[2];

  char* ws = (char*)d_ws;
  _Float16* WA   = (_Float16*)(ws);              // 128*128*2 = 32768 B
  _Float16* WB   = (_Float16*)(ws + 32768);      // 32768 B
  float*    wc   = (float*)(ws + 65536);         // 512 B
  float*    bias = (float*)(ws + 66048);         // 512 B

  prep_weights<<<64, 256, 0, stream>>>(W1, b1, W2, b2, WA, WB, wc, bias);

  const int blocks = (E + (TPB * 16) - 1) / (TPB * 16);
  edge_mlp<<<blocks, 256, 0, stream>>>(atom, bond, src, dst, WA, WB, wc, bias,
                                       out, E);
}
